// NaiveTransformerBlock_15857019257298
// MI455X (gfx1250) — compile-verified, matrix operand order checked by reading, not run on hardware
//
#include <hip/hip_runtime.h>
#include <hip/hip_bf16.h>
#include <hip/hip_fp16.h>

typedef __attribute__((ext_vector_type(16))) _Float16 v16h;
typedef __attribute__((ext_vector_type(8)))  float    v8f;

#define D_MODEL 1024
#define N_HEADS 16
#define D_K     64
#define D_FF    4096
#define BATCH   2
#define SEQ     2048
#define M_TOK   (BATCH*SEQ)
#define LN_EPS  1e-5f

// ---------------------------------------------------------------------------
// WMMA helpers
// ---------------------------------------------------------------------------
__device__ __forceinline__ v8f wmma_f16(v16h a, v16h b, v8f c) {
  // D = A(16x32 f16) * B(32x16 f16) + C(16x16 f32)
  return __builtin_amdgcn_wmma_f32_16x16x32_f16(
      /*neg_a=*/false, a, /*neg_b=*/false, b,
      /*c_mod=*/(short)0, c, /*reuse_a=*/false, /*reuse_b=*/false);
}

// Load an A/B fragment from a row-major f16 matrix.
// rowp = &mat[row * ld + k0] where row = lane%16 for A (rows) / B (cols of W[N,K]).
// Per-lane halves: [koff .. koff+7] and [16+koff .. 16+koff+7], koff = (lane>=16)*8.
__device__ __forceinline__ v16h load_frag(const _Float16* rowp, int koff) {
  union { v16h v; uint4 u[2]; } t;
  t.u[0] = *reinterpret_cast<const uint4*>(rowp + koff);
  t.u[1] = *reinterpret_cast<const uint4*>(rowp + 16 + koff);
  return t.v;
}

// ---------------------------------------------------------------------------
// fp32 -> fp16 conversion
// ---------------------------------------------------------------------------
__global__ __launch_bounds__(256) void cvt_f16_kernel(const float* __restrict__ in,
                                                      _Float16* __restrict__ out, int n) {
  int i = blockIdx.x * 256 + threadIdx.x;
  if (i < n) out[i] = (_Float16)in[i];
}

// ---------------------------------------------------------------------------
// LayerNorm: one block per row of D_MODEL, fp32 in, fp16 out
// ---------------------------------------------------------------------------
__global__ __launch_bounds__(256) void layernorm_kernel(const float* __restrict__ x,
                                                        const float* __restrict__ g,
                                                        const float* __restrict__ bta,
                                                        _Float16* __restrict__ out) {
  __shared__ float red[256];
  const int tid = threadIdx.x;
  const int row = blockIdx.x;
  const float* xr = x + (size_t)row * D_MODEL;

  float s = 0.f;
  for (int i = tid; i < D_MODEL; i += 256) s += xr[i];
  red[tid] = s; __syncthreads();
  for (int off = 128; off > 0; off >>= 1) {
    if (tid < off) red[tid] += red[tid + off];
    __syncthreads();
  }
  const float mu = red[0] * (1.0f / D_MODEL);
  __syncthreads();

  float vs = 0.f;
  for (int i = tid; i < D_MODEL; i += 256) { float d = xr[i] - mu; vs += d * d; }
  red[tid] = vs; __syncthreads();
  for (int off = 128; off > 0; off >>= 1) {
    if (tid < off) red[tid] += red[tid + off];
    __syncthreads();
  }
  const float rstd = rsqrtf(red[0] * (1.0f / D_MODEL) + LN_EPS);

  _Float16* orow = out + (size_t)row * D_MODEL;
  for (int i = tid; i < D_MODEL; i += 256)
    orow[i] = (_Float16)((xr[i] - mu) * rstd * g[i] + bta[i]);
}

// ---------------------------------------------------------------------------
// GEMM: C[M,N] = A[M,K] @ W[N,K]^T + bias (+relu) (+residual)
// One wave computes a 32(M) x 64(N) tile: 2 A-frags x 4 B-frags -> 8 WMMA per
// K-step against 12 b128 loads (load:WMMA = 1.5). A, W are f16 row-major.
// ---------------------------------------------------------------------------
__global__ __launch_bounds__(32)
void gemm_wmma_kernel(const _Float16* __restrict__ A, const _Float16* __restrict__ W,
                      const float* __restrict__ bias, const float* __restrict__ residual,
                      float* __restrict__ outF, _Float16* __restrict__ outH,
                      int M, int N, int K, int relu) {
  const int lane = threadIdx.x;
  const int lr   = lane & 15;
  const int lh   = lane >> 4;
  const int koff = lh * 8;

  const int mb = blockIdx.x * 32;
  const int nb = blockIdx.y * 64;

  const _Float16* arow0 = A + (size_t)(mb +      lr) * K;
  const _Float16* arow1 = A + (size_t)(mb + 16 + lr) * K;
  const _Float16* wrow0 = W + (size_t)(nb +      lr) * K;
  const _Float16* wrow1 = W + (size_t)(nb + 16 + lr) * K;
  const _Float16* wrow2 = W + (size_t)(nb + 32 + lr) * K;
  const _Float16* wrow3 = W + (size_t)(nb + 48 + lr) * K;

  v8f acc[2][4] = { { {}, {}, {}, {} }, { {}, {}, {}, {} } };

  for (int k0 = 0; k0 < K; k0 += 32) {
    __builtin_prefetch(arow0 + k0 + 64, 0, 1);
    __builtin_prefetch(wrow0 + k0 + 64, 0, 1);
    v16h a0 = load_frag(arow0 + k0, koff);
    v16h a1 = load_frag(arow1 + k0, koff);
    v16h b0 = load_frag(wrow0 + k0, koff);
    v16h b1 = load_frag(wrow1 + k0, koff);
    v16h b2 = load_frag(wrow2 + k0, koff);
    v16h b3 = load_frag(wrow3 + k0, koff);
    acc[0][0] = wmma_f16(a0, b0, acc[0][0]);
    acc[1][0] = wmma_f16(a1, b0, acc[1][0]);
    acc[0][1] = wmma_f16(a0, b1, acc[0][1]);
    acc[1][1] = wmma_f16(a1, b1, acc[1][1]);
    acc[0][2] = wmma_f16(a0, b2, acc[0][2]);
    acc[1][2] = wmma_f16(a1, b2, acc[1][2]);
    acc[0][3] = wmma_f16(a0, b3, acc[0][3]);
    acc[1][3] = wmma_f16(a1, b3, acc[1][3]);
  }

  const float bi[4] = { bias[nb + lr], bias[nb + 16 + lr],
                        bias[nb + 32 + lr], bias[nb + 48 + lr] };

#pragma unroll
  for (int r = 0; r < 2; ++r) {
#pragma unroll
    for (int v = 0; v < 8; ++v) {
      const int m = mb + r * 16 + lh * 8 + v;   // D layout: row = vgpr + (lane>=16)*8
      const size_t ro = (size_t)m * N;
#pragma unroll
      for (int t = 0; t < 4; ++t) {
        const int n = nb + t * 16 + lr;         // D layout: col = lane%16
        float val = acc[r][t][v] + bi[t];
        if (relu) val = fmaxf(val, 0.0f);
        if (residual) val += residual[ro + n];
        if (outF) outF[ro + n] = val;
        if (outH) outH[ro + n] = (_Float16)val;
      }
    }
  }
}

// ---------------------------------------------------------------------------
// Flash-style attention. One wave per (batch, head, 16-query block).
// Q,K,V are f16 [B, S, H*D_K] row-major (head h occupies cols h*64..h*64+63).
// 32 keys per iteration: 4 WMMA for scores, online softmax, 4 WMMA for P@V.
// ---------------------------------------------------------------------------
__global__ __launch_bounds__(32)
void attn_wmma_kernel(const _Float16* __restrict__ Q, const _Float16* __restrict__ Kt,
                      const _Float16* __restrict__ V, _Float16* __restrict__ O) {
  __shared__ _Float16 lds[16 * 40];          // P tile 16x32, row stride 40 (16B aligned)

  const int lane = threadIdx.x;
  const int lr   = lane & 15;
  const int lh   = lane >> 4;
  const int koff = lh * 8;

  const int qb = blockIdx.x & 127;           // 128 query blocks of 16
  const int h  = (blockIdx.x >> 7) & 15;
  const int b  = blockIdx.x >> 11;

  const _Float16* qrow = Q + ((size_t)(b * SEQ + qb * 16 + lr) * D_MODEL) + h * D_K;
  const v16h aq0 = load_frag(qrow,      koff);   // dims 0..31
  const v16h aq1 = load_frag(qrow + 32, koff);   // dims 32..63

  const _Float16* kbase = Kt + (size_t)b * SEQ * D_MODEL + h * D_K;
  const _Float16* vbase = V  + (size_t)b * SEQ * D_MODEL + h * D_K;

  v8f oacc[4] = { {}, {}, {}, {} };
  float rmax[8], rsum[8];
#pragma unroll
  for (int v = 0; v < 8; ++v) { rmax[v] = -1e30f; rsum[v] = 0.0f; }

  for (int s0 = 0; s0 < SEQ; s0 += 32) {
    // ---- scores: 16 queries x 32 keys -----------------------------------
    const _Float16* krow0 = kbase + (size_t)(s0 +      lr) * D_MODEL;
    const _Float16* krow1 = kbase + (size_t)(s0 + 16 + lr) * D_MODEL;
    v8f sc0 = {}, sc1 = {};
    sc0 = wmma_f16(aq0, load_frag(krow0,      koff), sc0);
    sc0 = wmma_f16(aq1, load_frag(krow0 + 32, koff), sc0);
    sc1 = wmma_f16(aq0, load_frag(krow1,      koff), sc1);
    sc1 = wmma_f16(aq1, load_frag(krow1 + 32, koff), sc1);

    // ---- online softmax over each query row (16 lanes per half own a row)
#pragma unroll
    for (int v = 0; v < 8; ++v) {
      float s0v = sc0[v] * 0.125f;           // 1/sqrt(64)
      float s1v = sc1[v] * 0.125f;
      float mx = fmaxf(s0v, s1v);
      mx = fmaxf(mx, __shfl_xor(mx, 1));
      mx = fmaxf(mx, __shfl_xor(mx, 2));
      mx = fmaxf(mx, __shfl_xor(mx, 4));
      mx = fmaxf(mx, __shfl_xor(mx, 8));
      const float mnew  = fmaxf(rmax[v], mx);
      const float alpha = __expf(rmax[v] - mnew);
      const float p0 = __expf(s0v - mnew);
      const float p1 = __expf(s1v - mnew);
      float ps = p0 + p1;
      ps += __shfl_xor(ps, 1);
      ps += __shfl_xor(ps, 2);
      ps += __shfl_xor(ps, 4);
      ps += __shfl_xor(ps, 8);
      rsum[v] = rsum[v] * alpha + ps;
      rmax[v] = mnew;
      oacc[0][v] *= alpha; oacc[1][v] *= alpha;
      oacc[2][v] *= alpha; oacc[3][v] *= alpha;
      const int prow = lh * 8 + v;           // D-layout row this lane owns
      lds[prow * 40 +      lr] = (_Float16)p0;   // keys  0..15
      lds[prow * 40 + 16 + lr] = (_Float16)p1;   // keys 16..31
    }
    __syncthreads();

    // ---- reload P as an A-fragment (16 rows x 32 keys) -------------------
    const v16h ap = load_frag(&lds[lr * 40], koff);

    // ---- O += P(16x32) @ V(32x64) : B-frag cols = head dims --------------
#pragma unroll
    for (int t = 0; t < 4; ++t) {
      union { v16h v; _Float16 e[16]; } bv;
#pragma unroll
      for (int e = 0; e < 16; ++e) {
        const int key = (e < 8) ? (koff + e) : (16 + koff + (e - 8));
        bv.e[e] = vbase[(size_t)(s0 + key) * D_MODEL + t * 16 + lr];
      }
      oacc[t] = wmma_f16(ap, bv.v, oacc[t]);
    }
    __syncthreads();
  }

  // ---- normalize and store (heads concatenate naturally in D_MODEL) -----
#pragma unroll
  for (int v = 0; v < 8; ++v) {
    const float inv = 1.0f / rsum[v];
    const int m = qb * 16 + lh * 8 + v;
    _Float16* orow = O + (size_t)(b * SEQ + m) * D_MODEL + h * D_K;
    orow[     lr] = (_Float16)(oacc[0][v] * inv);
    orow[16 + lr] = (_Float16)(oacc[1][v] * inv);
    orow[32 + lr] = (_Float16)(oacc[2][v] * inv);
    orow[48 + lr] = (_Float16)(oacc[3][v] * inv);
  }
}

// ---------------------------------------------------------------------------
// Orchestration
// ---------------------------------------------------------------------------
extern "C" void kernel_launch(void* const* d_in, const int* in_sizes, int n_in,
                              void* d_out, int out_size, void* d_ws, size_t ws_size,
                              hipStream_t stream) {
  (void)in_sizes; (void)n_in; (void)out_size; (void)ws_size;

  const float* x     = (const float*)d_in[0];
  const float* ln1g  = (const float*)d_in[1];
  const float* ln1b  = (const float*)d_in[2];
  const float* wq    = (const float*)d_in[3];
  const float* wqb   = (const float*)d_in[4];
  const float* wk    = (const float*)d_in[5];
  const float* wkb   = (const float*)d_in[6];
  const float* wv    = (const float*)d_in[7];
  const float* wvb   = (const float*)d_in[8];
  const float* wo    = (const float*)d_in[9];
  const float* wob   = (const float*)d_in[10];
  const float* ln2g  = (const float*)d_in[11];
  const float* ln2b  = (const float*)d_in[12];
  const float* w1    = (const float*)d_in[13];
  const float* w1b   = (const float*)d_in[14];
  const float* w2    = (const float*)d_in[15];
  const float* w2b   = (const float*)d_in[16];
  float* out = (float*)d_out;

  char* ws = (char*)d_ws;
  const size_t MB = 1ull << 20;
  _Float16* wq_h = (_Float16*)(ws +  0 * MB);   // 2 MB
  _Float16* wk_h = (_Float16*)(ws +  2 * MB);   // 2 MB
  _Float16* wv_h = (_Float16*)(ws +  4 * MB);   // 2 MB
  _Float16* wo_h = (_Float16*)(ws +  6 * MB);   // 2 MB
  _Float16* w1_h = (_Float16*)(ws +  8 * MB);   // 8 MB
  _Float16* w2_h = (_Float16*)(ws + 16 * MB);   // 8 MB
  _Float16* h1   = (_Float16*)(ws + 24 * MB);   // 8 MB   LN1 output
  _Float16* qh   = (_Float16*)(ws + 32 * MB);   // 8 MB
  _Float16* kh   = (_Float16*)(ws + 40 * MB);   // 8 MB
  _Float16* vh   = (_Float16*)(ws + 48 * MB);   // 8 MB
  _Float16* attn = (_Float16*)(ws + 56 * MB);   // 8 MB
  float*    x2   = (float*)   (ws + 64 * MB);   // 16 MB  post-attn residual
  _Float16* h2   = (_Float16*)(ws + 80 * MB);   // 8 MB   LN2 output
  _Float16* ff   = (_Float16*)(ws + 88 * MB);   // 32 MB  FFN hidden

  const dim3 b256(256), b32(32);

  // 1) weights -> f16
  cvt_f16_kernel<<<(D_MODEL * D_MODEL + 255) / 256, b256, 0, stream>>>(wq, wq_h, D_MODEL * D_MODEL);
  cvt_f16_kernel<<<(D_MODEL * D_MODEL + 255) / 256, b256, 0, stream>>>(wk, wk_h, D_MODEL * D_MODEL);
  cvt_f16_kernel<<<(D_MODEL * D_MODEL + 255) / 256, b256, 0, stream>>>(wv, wv_h, D_MODEL * D_MODEL);
  cvt_f16_kernel<<<(D_MODEL * D_MODEL + 255) / 256, b256, 0, stream>>>(wo, wo_h, D_MODEL * D_MODEL);
  cvt_f16_kernel<<<(D_FF * D_MODEL + 255) / 256,   b256, 0, stream>>>(w1, w1_h, D_FF * D_MODEL);
  cvt_f16_kernel<<<(D_MODEL * D_FF + 255) / 256,   b256, 0, stream>>>(w2, w2_h, D_MODEL * D_FF);

  // 2) LN1
  layernorm_kernel<<<M_TOK, b256, 0, stream>>>(x, ln1g, ln1b, h1);

  // 3) Q/K/V projections (f16 out, row-major [tok, H*D_K])
  {
    dim3 grid(M_TOK / 32, D_MODEL / 64);
    gemm_wmma_kernel<<<grid, b32, 0, stream>>>(h1, wq_h, wqb, nullptr, nullptr, qh,
                                               M_TOK, D_MODEL, D_MODEL, 0);
    gemm_wmma_kernel<<<grid, b32, 0, stream>>>(h1, wk_h, wkb, nullptr, nullptr, kh,
                                               M_TOK, D_MODEL, D_MODEL, 0);
    gemm_wmma_kernel<<<grid, b32, 0, stream>>>(h1, wv_h, wvb, nullptr, nullptr, vh,
                                               M_TOK, D_MODEL, D_MODEL, 0);
  }

  // 4) attention: B*H*(S/16) = 4096 waves
  attn_wmma_kernel<<<BATCH * N_HEADS * (SEQ / 16), b32, 0, stream>>>(qh, kh, vh, attn);

  // 5) output projection + residual -> x2 (fp32)
  {
    dim3 grid(M_TOK / 32, D_MODEL / 64);
    gemm_wmma_kernel<<<grid, b32, 0, stream>>>(attn, wo_h, wob, x, x2, nullptr,
                                               M_TOK, D_MODEL, D_MODEL, 0);
  }

  // 6) LN2
  layernorm_kernel<<<M_TOK, b256, 0, stream>>>(x2, ln2g, ln2b, h2);

  // 7) FFN1 + ReLU (f16 out)
  {
    dim3 grid(M_TOK / 32, D_FF / 64);
    gemm_wmma_kernel<<<grid, b32, 0, stream>>>(h2, w1_h, w1b, nullptr, nullptr, ff,
                                               M_TOK, D_FF, D_MODEL, 1);
  }

  // 8) FFN2 + residual -> d_out (fp32)
  {
    dim3 grid(M_TOK / 32, D_MODEL / 64);
    gemm_wmma_kernel<<<grid, b32, 0, stream>>>(ff, w2_h, w2b, x2, out, nullptr,
                                               M_TOK, D_MODEL, D_FF, 0);
  }
}
